// GATEncoder_30872224924147
// MI455X (gfx1250) — matrix operand order repaired and checked by reading, not compiled, not run on hardware
//
#include <hip/hip_runtime.h>
#include <math.h>

#define N_NODES   8192
#define IN_DIM    512
#define H_HEADS   8
#define C_DIM     128
#define HC_DIM    1024
#define E_EDGES   131072
#define ETOT      (E_EDGES + N_NODES)   /* 139264 edges incl. self-loops */
#define BN_EPS    1e-5f
#define NEG_SLOPE 0.2f

typedef __attribute__((ext_vector_type(16))) __bf16 v16bf;
typedef __attribute__((ext_vector_type(8)))  float  v8f;

// ---------------------------------------------------------------------------
// Pack activations A[M,K] (fp32, row-major) into WMMA A-fragment layout:
//   frag index f = mt*(K/32) + ks ; element (f, lane) holds 16 bf16:
//   row = 16*mt + (lane&15), kh = 8*(lane>=16),
//   i<8  -> k = 32*ks + kh + i ; i>=8 -> k = 32*ks + 16 + kh + (i-8)
// One thread writes one 32-byte per-lane fragment slice (coalesced).
// ---------------------------------------------------------------------------
__global__ __launch_bounds__(256) void pack_a(
    const float* __restrict__ A, v16bf* __restrict__ Ap, int M, int K)
{
  int tid = blockIdx.x * 256 + threadIdx.x;
  int KS = K >> 5;
  int total = (M >> 4) * KS * 32;
  if (tid >= total) return;
  int lane = tid & 31;
  int frag = tid >> 5;
  int ks = frag % KS;
  int mt = frag / KS;
  int row = mt * 16 + (lane & 15);
  int kh  = (lane >> 4) << 3;
  const float* p = A + (size_t)row * K + ks * 32 + kh;
  v16bf v;
#pragma unroll
  for (int i = 0; i < 8; ++i) v[i] = (__bf16)p[i];
#pragma unroll
  for (int i = 0; i < 8; ++i) v[8 + i] = (__bf16)p[16 + i];
  Ap[tid] = v;
}

// ---------------------------------------------------------------------------
// Pack weights W[K,Nc] (fp32, row-major) into WMMA B-fragment layout:
//   frag f = nt*(K/32) + ks ; (f, lane): n = 16*nt + (lane&15),
//   kb = 8*(lane>=16), i<8 -> k = 32*ks + kb + i ; i>=8 -> k = 32*ks + 16 + kb + (i-8).
// ---------------------------------------------------------------------------
__global__ __launch_bounds__(256) void pack_w(
    const float* __restrict__ W, v16bf* __restrict__ Wp, int K, int Nc)
{
  int tid = blockIdx.x * 256 + threadIdx.x;
  int KS = K >> 5;
  int total = (Nc >> 4) * KS * 32;
  if (tid >= total) return;
  int lane = tid & 31;
  int frag = tid >> 5;
  int ks = frag % KS;
  int nt = frag / KS;
  int n  = nt * 16 + (lane & 15);
  int kb = (lane >> 4) << 3;
  const float* p = W + (size_t)(ks * 32 + kb) * Nc + n;
  v16bf v;
#pragma unroll
  for (int i = 0; i < 16; ++i) v[i] = (__bf16)p[(size_t)(i + (i & 8)) * Nc];
  Wp[tid] = v;
}

// ---------------------------------------------------------------------------
// GEMM on packed fragments: C[M,Nc] = A @ W + bias (bf16 WMMA, fp32 accum).
// Wave -> 32 rows x 64 cols (2 m-tiles x 4 n-tiles, 8 WMMAs / K-step,
// B fragments reused across both m-tiles). 4 waves/block -> 128x64.
// Inner loop: 12 x b128 loads : 8 x v_wmma. No divergence; EXEC all-ones.
// ---------------------------------------------------------------------------
__global__ __launch_bounds__(128) void gemm_packed_wmma(
    const v16bf* __restrict__ Ap, const v16bf* __restrict__ Wp,
    const float* __restrict__ bias, float* __restrict__ C,
    int M, int K, int Nc)
{
  const int lane = threadIdx.x & 31;
  const int wave = threadIdx.x >> 5;
  const int KS = K >> 5;
  const int mt0 = blockIdx.x * 8 + wave * 2;   // first of 2 m-tiles
  const int nt0 = blockIdx.y * 4;              // first of 4 n-tiles

  const v16bf* a0p = Ap + (size_t)(mt0 + 0) * KS * 32 + lane;
  const v16bf* a1p = Ap + (size_t)(mt0 + 1) * KS * 32 + lane;
  const v16bf* wp  = Wp + (size_t)nt0 * KS * 32 + lane;
  const size_t ntstride = (size_t)KS * 32;

  v8f acc[2][4] = {};

  for (int ks = 0; ks < KS; ++ks) {
    v16bf a0 = a0p[(size_t)ks * 32];
    v16bf a1 = a1p[(size_t)ks * 32];
#pragma unroll
    for (int t = 0; t < 4; ++t) {
      v16bf b = wp[(size_t)t * ntstride + (size_t)ks * 32];
      acc[0][t] = __builtin_amdgcn_wmma_f32_16x16x32_bf16(
          false, a0, false, b, (short)0, acc[0][t], false, false);
      acc[1][t] = __builtin_amdgcn_wmma_f32_16x16x32_bf16(
          false, a1, false, b, (short)0, acc[1][t], false, false);
    }
  }

  // D layout: element v of lane -> m = 8*(lane>=16) + v, n = lane&15
#pragma unroll
  for (int r = 0; r < 2; ++r) {
    int mbase = (mt0 + r) * 16 + ((lane >> 4) << 3);
#pragma unroll
    for (int t = 0; t < 4; ++t) {
      int n = (nt0 + t) * 16 + (lane & 15);
      float bv = bias[n];
#pragma unroll
      for (int v = 0; v < 8; ++v) {
        C[(size_t)(mbase + v) * Nc + n] = acc[r][t][v] + bv;
      }
    }
  }
}

// ---------------------------------------------------------------------------
// Order-preserving float <-> uint encoding for atomicMax-based segment max
// ---------------------------------------------------------------------------
__device__ inline unsigned enc_ord(float f) {
  unsigned u = __float_as_uint(f);
  return (u & 0x80000000u) ? ~u : (u | 0x80000000u);
}
__device__ inline float dec_ord(unsigned e) {
  unsigned u = (e & 0x80000000u) ? (e & 0x7fffffffu) : ~e;
  return __uint_as_float(u);
}
__device__ inline float lrelu(float v) { return v > 0.f ? v : NEG_SLOPE * v; }

__device__ inline void edge_ends(const int* __restrict__ ei, int e, int& src, int& dst) {
  if (e < E_EDGES) { src = ei[e]; dst = ei[E_EDGES + e]; }
  else             { src = dst = e - E_EDGES; }          // self-loop
}

// emax -> enc(-inf), denom -> 0, column stats -> 0
__global__ void init_edge_state(unsigned* __restrict__ emax, float* __restrict__ denom,
                                float* __restrict__ sum, float* __restrict__ sumsq)
{
  int t = blockIdx.x * blockDim.x + threadIdx.x;
  if (t < N_NODES * H_HEADS) { emax[t] = 0x007FFFFFu; denom[t] = 0.f; }
  if (t < HC_DIM)            { sum[t] = 0.f; sumsq[t] = 0.f; }
}

__global__ void zero_f4(float4* __restrict__ p, int n4)
{
  int t = blockIdx.x * blockDim.x + threadIdx.x;
  if (t < n4) p[t] = make_float4(0.f, 0.f, 0.f, 0.f);
}

// Per-(edge,head) attention logit + segment max over dst
__global__ __launch_bounds__(256) void edge_logits(
    const float* __restrict__ xl, const float* __restrict__ xr,
    const int* __restrict__ ei, const float* __restrict__ att,
    float* __restrict__ scores, unsigned* __restrict__ emax)
{
  int t = blockIdx.x * 256 + threadIdx.x;
  if (t >= ETOT * H_HEADS) return;
  int e = t >> 3, h = t & 7;
  int src, dst; edge_ends(ei, e, src, dst);

  const float4* pl = (const float4*)(xl + (size_t)src * HC_DIM + h * C_DIM);
  const float4* pr = (const float4*)(xr + (size_t)dst * HC_DIM + h * C_DIM);
  const float4* pa = (const float4*)(att + h * C_DIM);
  float s = 0.f;
#pragma unroll 8
  for (int i = 0; i < C_DIM / 4; ++i) {
    float4 l = pl[i], r = pr[i], a = pa[i];
    s += lrelu(l.x + r.x) * a.x;
    s += lrelu(l.y + r.y) * a.y;
    s += lrelu(l.z + r.z) * a.z;
    s += lrelu(l.w + r.w) * a.w;
  }
  scores[t] = s;
  atomicMax(emax + dst * H_HEADS + h, enc_ord(s));
}

// w = exp(e - emax[dst]); denom[dst] += w
__global__ __launch_bounds__(256) void edge_softmax_num(
    const int* __restrict__ ei, float* __restrict__ scores,
    const unsigned* __restrict__ emax, float* __restrict__ denom)
{
  int t = blockIdx.x * 256 + threadIdx.x;
  if (t >= ETOT * H_HEADS) return;
  int e = t >> 3, h = t & 7;
  int src, dst; edge_ends(ei, e, src, dst);
  float w = expf(scores[t] - dec_ord(emax[dst * H_HEADS + h]));
  scores[t] = w;
  atomicAdd(denom + dst * H_HEADS + h, w);
}

// agg[dst] += (w / denom[dst]) * xl[src]   (thread per edge x float4-chunk)
__global__ __launch_bounds__(256) void edge_aggregate(
    const float* __restrict__ xl, const int* __restrict__ ei,
    const float* __restrict__ scores, const float* __restrict__ denom,
    float* __restrict__ agg)
{
  int t = blockIdx.x * 256 + threadIdx.x;
  if (t >= ETOT * (HC_DIM / 4)) return;
  int e  = t >> 8;          // / (HC/4)
  int c4 = t & 255;
  int h  = c4 >> 5;         // c4*4 / 128
  int src, dst; edge_ends(ei, e, src, dst);

  float alpha = scores[e * H_HEADS + h] / denom[dst * H_HEADS + h];
  float4 v = ((const float4*)(xl + (size_t)src * HC_DIM))[c4];
  float* o = agg + (size_t)dst * HC_DIM + c4 * 4;
  atomicAdd(o + 0, alpha * v.x);
  atomicAdd(o + 1, alpha * v.y);
  atomicAdd(o + 2, alpha * v.z);
  atomicAdd(o + 3, alpha * v.w);
}

// Column sums for BN: one column per thread (coalesced), 32 row-slabs
__global__ __launch_bounds__(256) void bn_stats(
    const float* __restrict__ x, float* __restrict__ sum, float* __restrict__ sumsq)
{
  int col = blockIdx.x * 256 + threadIdx.x;          // HC/256 blocks in x
  int r0  = blockIdx.y * (N_NODES / 32);
  float s = 0.f, s2 = 0.f;
  for (int r = r0; r < r0 + N_NODES / 32; ++r) {
    float v = x[(size_t)r * HC_DIM + col];
    s += v; s2 += v * v;
  }
  atomicAdd(sum + col, s);
  atomicAdd(sumsq + col, s2);
}

// y = elu( (x - mean) * rsqrt(var+eps) * g + beta )
__global__ __launch_bounds__(256) void bn_elu(
    const float* __restrict__ x, const float* __restrict__ sum,
    const float* __restrict__ sumsq, const float* __restrict__ g,
    const float* __restrict__ beta, float* __restrict__ out)
{
  int t = blockIdx.x * 256 + threadIdx.x;
  if (t >= N_NODES * HC_DIM) return;
  int col = t & (HC_DIM - 1);
  const float invN = 1.f / (float)N_NODES;
  float mean = sum[col] * invN;
  float var  = sumsq[col] * invN - mean * mean;
  float y = (x[t] - mean) * rsqrtf(var + BN_EPS) * g[col] + beta[col];
  out[t] = y > 0.f ? y : expm1f(y);
}

// ---------------------------------------------------------------------------
extern "C" void kernel_launch(void* const* d_in, const int* in_sizes, int n_in,
                              void* d_out, int out_size, void* d_ws, size_t ws_size,
                              hipStream_t stream)
{
  const float* x    = (const float*)d_in[0];
  const int*   ei   = (const int*)  d_in[1];
  const float* W1l  = (const float*)d_in[2];
  const float* b1l  = (const float*)d_in[3];
  const float* W1r  = (const float*)d_in[4];
  const float* b1r  = (const float*)d_in[5];
  const float* att1 = (const float*)d_in[6];
  /* bias1 (d_in[7]) absorbed by BN mean-subtraction */
  const float* g1   = (const float*)d_in[8];
  const float* bt1  = (const float*)d_in[9];
  const float* W2l  = (const float*)d_in[10];
  const float* b2l  = (const float*)d_in[11];
  const float* W2r  = (const float*)d_in[12];
  const float* b2r  = (const float*)d_in[13];
  const float* att2 = (const float*)d_in[14];
  /* bias2 (d_in[15]) absorbed by BN */
  const float* g2   = (const float*)d_in[16];
  const float* bt2  = (const float*)d_in[17];
  float* out = (float*)d_out;

  // ---- Workspace layout (~93 MB) ----
  const size_t NHC = (size_t)N_NODES * HC_DIM;
  float*    ws     = (float*)d_ws;
  float*    xl     = ws;                      // [N,HC] fp32
  float*    xr     = ws + NHC;                // [N,HC]; reused as agg after logits
  float*    agg    = xr;
  float*    scores = ws + 2 * NHC;            // [ETOT,H] fp32
  unsigned* emax   = (unsigned*)(scores + (size_t)ETOT * H_HEADS); // [N,H]
  float*    denom  = (float*)(emax + (size_t)N_NODES * H_HEADS);   // [N,H]
  float*    sum    = denom + (size_t)N_NODES * H_HEADS;            // [HC]
  float*    sumsq  = sum + HC_DIM;                                 // [HC]
  // packed bf16 buffers (32B-aligned offsets)
  v16bf*    Ap     = (v16bf*)(sumsq + HC_DIM);                     // up to M*K bf16 (16 MB)
  v16bf*    Wlp    = (v16bf*)((char*)Ap  + (size_t)N_NODES * HC_DIM * 2); // K*Nc bf16 (2 MB)
  v16bf*    Wrp    = (v16bf*)((char*)Wlp + (size_t)HC_DIM * HC_DIM * 2);  // K*Nc bf16 (2 MB)

  const int eh_blocks  = (ETOT * H_HEADS + 255) / 256;
  const int agg_blocks = (ETOT * (HC_DIM / 4) + 255) / 256;
  const int nhc_blocks = (int)((NHC + 255) / 256);
  const dim3 gemm_grid(N_NODES / 128, HC_DIM / 64);

  auto run_layer = [&](const float* in, int K,
                       const float* Wl, const float* bl,
                       const float* Wr, const float* br,
                       const float* att, const float* g, const float* beta,
                       float* layer_out) {
    const int KS = K / 32;
    const int a_threads = (N_NODES / 16) * KS * 32;
    const int w_threads = (HC_DIM / 16) * KS * 32;
    pack_a<<<(a_threads + 255) / 256, 256, 0, stream>>>(in, Ap, N_NODES, K);
    pack_w<<<(w_threads + 255) / 256, 256, 0, stream>>>(Wl, Wlp, K, HC_DIM);
    pack_w<<<(w_threads + 255) / 256, 256, 0, stream>>>(Wr, Wrp, K, HC_DIM);
    gemm_packed_wmma<<<gemm_grid, 128, 0, stream>>>(Ap, Wlp, bl, xl, N_NODES, K, HC_DIM);
    gemm_packed_wmma<<<gemm_grid, 128, 0, stream>>>(Ap, Wrp, br, xr, N_NODES, K, HC_DIM);
    init_edge_state<<<(N_NODES * H_HEADS + 255) / 256, 256, 0, stream>>>(emax, denom, sum, sumsq);
    edge_logits<<<eh_blocks, 256, 0, stream>>>(xl, xr, ei, att, scores, emax);
    // xr is dead now; zero it for use as the aggregation buffer
    zero_f4<<<(int)((NHC / 4 + 255) / 256), 256, 0, stream>>>((float4*)agg, (int)(NHC / 4));
    edge_softmax_num<<<eh_blocks, 256, 0, stream>>>(ei, scores, emax, denom);
    edge_aggregate<<<agg_blocks, 256, 0, stream>>>(xl, ei, scores, denom, agg);
    bn_stats<<<dim3(HC_DIM / 256, 32), 256, 0, stream>>>(agg, sum, sumsq);
    bn_elu<<<nhc_blocks, 256, 0, stream>>>(agg, sum, sumsq, g, beta, layer_out);
  };

  // Layer 1: input x (K=512); hidden h lands in d_out (fully overwritten).
  run_layer(x, IN_DIM, W1l, b1l, W1r, b1r, att1, g1, bt1, out);
  // Layer 2: input h = d_out (K=1024); final result overwrites d_out.
  run_layer(out, HC_DIM, W2l, b2l, W2r, b2r, att2, g2, bt2, out);

  (void)in_sizes; (void)n_in; (void)out_size; (void)ws_size;
}
